// SharedGRUUnroll_75539884802374
// MI455X (gfx1250) — compile-verified, matrix operand order checked by reading, not run on hardware
//
#include <hip/hip_runtime.h>

// ---------------------------------------------------------------------------
// GRU (reset_after=True), B=256, T=256, I=512, U=1024 on MI455X (gfx1250).
// LDS-free step kernel: bf16 WMMA 16x16x32 fragments loaded directly from
// L2-resident pre-transposed bf16 weights; h ping-ponged as f32 + bf16.
// ---------------------------------------------------------------------------

typedef __attribute__((ext_vector_type(16))) __bf16 v16bf;
typedef __attribute__((ext_vector_type(8)))  float  v8f;
typedef unsigned short u16;

#define B_SZ   256
#define T_SZ   256
#define I_SZ   512
#define U_SZ   1024
#define G3     (3 * U_SZ)          // 3072 gate columns

#define WG_UNITS 16                // hidden units per workgroup (N tile)
#define WG_ROWS  128               // batch rows per workgroup (8 waves x 16)

struct FragBits { uint4 lo; uint4 hi; };   // 32 bytes == v16bf

__device__ __forceinline__ float sigmoid_f(float x) {
    return 1.0f / (1.0f + __expf(-x));
}

__device__ __forceinline__ float tanh_f(float x) {
    x = fminf(fmaxf(x, -15.0f), 15.0f);
    float e = __expf(-2.0f * x);
    return (1.0f - e) / (1.0f + e);
}

__device__ __forceinline__ u16 bf_bits(float f) {
    return __builtin_bit_cast(u16, (__bf16)f);
}

// ---- prep: zero a u32 region ----
__global__ void gru_zero_u32_75539884802374(unsigned int* __restrict__ p, int n) {
    int i = blockIdx.x * blockDim.x + threadIdx.x;
    if (i < n) p[i] = 0u;
}

// ---- prep: f32 [K][3072] -> bf16 transposed [3072][K] ----
__global__ void gru_cvt_tr_75539884802374(const float* __restrict__ src,
                                          u16* __restrict__ dst, int K) {
    long long i = (long long)blockIdx.x * blockDim.x + threadIdx.x;  // < 3072*K
    int col = (int)(i / K);
    int k   = (int)(i % K);
    dst[(long long)col * K + k] = bf_bits(src[(long long)k * G3 + col]);
}

// ---------------------------------------------------------------------------
// One GRU timestep. 256 threads = 8 independent waves (no LDS, no barriers).
// Each wave: 16 rows x 16 units x 3 gate blocks -> 4 accumulators.
// ---------------------------------------------------------------------------
__global__ __launch_bounds__(256)
void gru_step_75539884802374(const float* __restrict__ xt,      // x + t*I, row stride T*I
                             const u16*   __restrict__ Wt,      // bf16 W^T  [3072][512]
                             const u16*   __restrict__ Ut,      // bf16 U^T  [3072][1024]
                             const float* __restrict__ bias,    // (2 x 3U)
                             const float* __restrict__ hprev,   // f32  (B x U)
                             const u16*   __restrict__ hprevb,  // bf16 (B x U)
                             float*       __restrict__ hnext,   // f32  (B x U)
                             u16*         __restrict__ hnextb)  // bf16 (B x U)
{
    const int tid  = threadIdx.x;
    const int lane = tid & 31;
    const int wave = tid >> 5;          // 0..7
    const int hsel = lane >> 4;         // WMMA K-interleave half
    const int lr   = lane & 15;
    const int u0   = blockIdx.x * WG_UNITS;
    const int rb   = blockIdx.y * WG_ROWS;
    const int arow = rb + wave * 16 + lr;   // A-matrix row this lane serves

    v8f acc0 = {};   // z gate
    v8f acc1 = {};   // r gate
    v8f acc2 = {};   // hh gate (per-phase part)
    v8f accx = {};   // saved x-part of hh gate
    const v8f vzero = {};

    // ---------------- phase 0: x_t @ W  (K = 512, f32 source) ----------------
    {
        const float* ap = xt + (long long)arow * (T_SZ * I_SZ);
        for (int k0 = 0; k0 < I_SZ; k0 += 32) {
            const int kb = k0 + hsel * 8;
            const float4 fa = *(const float4*)(ap + kb);
            const float4 fb = *(const float4*)(ap + kb + 4);
            const float4 fc = *(const float4*)(ap + kb + 16);
            const float4 fd = *(const float4*)(ap + kb + 20);
            v16bf a;
            a[0]  = (__bf16)fa.x; a[1]  = (__bf16)fa.y; a[2]  = (__bf16)fa.z; a[3]  = (__bf16)fa.w;
            a[4]  = (__bf16)fb.x; a[5]  = (__bf16)fb.y; a[6]  = (__bf16)fb.z; a[7]  = (__bf16)fb.w;
            a[8]  = (__bf16)fc.x; a[9]  = (__bf16)fc.y; a[10] = (__bf16)fc.z; a[11] = (__bf16)fc.w;
            a[12] = (__bf16)fd.x; a[13] = (__bf16)fd.y; a[14] = (__bf16)fd.z; a[15] = (__bf16)fd.w;

            #pragma unroll
            for (int g = 0; g < 3; ++g) {
                const int col = g * U_SZ + u0 + lr;                  // gate column
                const u16* bp = Wt + (long long)col * I_SZ + kb;
                FragBits pb;
                pb.lo = *(const uint4*)(bp);
                pb.hi = *(const uint4*)(bp + 16);
                const v16bf b = __builtin_bit_cast(v16bf, pb);
                v8f* accp = (g == 0) ? &acc0 : (g == 1) ? &acc1 : &acc2;
                *accp = __builtin_amdgcn_wmma_f32_16x16x32_bf16(
                            false, a, false, b, (short)0, *accp, false, false);
            }
        }
        accx = acc2;        // snapshot x-part of hh gate
        acc2 = vzero;
    }

    // ---------------- phase 1: h @ U  (K = 1024, bf16 source) ----------------
    {
        const u16* ap = hprevb + (long long)arow * U_SZ;
        for (int k0 = 0; k0 < U_SZ; k0 += 32) {
            const int kb = k0 + hsel * 8;
            FragBits pa;
            pa.lo = *(const uint4*)(ap + kb);
            pa.hi = *(const uint4*)(ap + kb + 16);
            const v16bf a = __builtin_bit_cast(v16bf, pa);

            #pragma unroll
            for (int g = 0; g < 3; ++g) {
                const int col = g * U_SZ + u0 + lr;
                const u16* bp = Ut + (long long)col * U_SZ + kb;
                FragBits pb;
                pb.lo = *(const uint4*)(bp);
                pb.hi = *(const uint4*)(bp + 16);
                const v16bf b = __builtin_bit_cast(v16bf, pb);
                v8f* accp = (g == 0) ? &acc0 : (g == 1) ? &acc1 : &acc2;
                *accp = __builtin_amdgcn_wmma_f32_16x16x32_bf16(
                            false, a, false, b, (short)0, *accp, false, false);
            }
        }
    }

    // ---------------- elementwise GRU update + dual writeback ----------------
    const int uu = u0 + lr;
    const float bz  = bias[uu]              + bias[G3 + uu];
    const float br  = bias[U_SZ + uu]       + bias[G3 + U_SZ + uu];
    const float bxh = bias[2 * U_SZ + uu];
    const float brh = bias[G3 + 2 * U_SZ + uu];

    #pragma unroll
    for (int j = 0; j < 8; ++j) {
        // C/D layout: VGPR j holds M=j (lanes 0-15) / M=j+8 (lanes 16-31), N=lane%16
        const int row = rb + wave * 16 + hsel * 8 + j;
        const float z  = sigmoid_f(acc0[j] + bz);
        const float r  = sigmoid_f(acc1[j] + br);
        const float hh = tanh_f(accx[j] + bxh + r * (acc2[j] + brh));
        const float hp = hprev[(long long)row * U_SZ + uu];
        const float hn = z * hp + (1.0f - z) * hh;
        hnext[(long long)row * U_SZ + uu]  = hn;
        hnextb[(long long)row * U_SZ + uu] = bf_bits(hn);
    }
}

extern "C" void kernel_launch(void* const* d_in, const int* in_sizes, int n_in,
                              void* d_out, int out_size, void* d_ws, size_t ws_size,
                              hipStream_t stream) {
    (void)in_sizes; (void)n_in; (void)out_size; (void)ws_size;

    const float* x    = (const float*)d_in[0];   // (B, T, I)
    const float* wk   = (const float*)d_in[1];   // (I, 3U)
    const float* rk   = (const float*)d_in[2];   // (U, 3U)
    const float* bias = (const float*)d_in[3];   // (2, 3U)
    float* out = (float*)d_out;                  // (B, U)

    // ws layout (12 MB total):
    //   [0)      h0 f32   : 1 MB
    //   [1 MB)   h1 f32   : 1 MB
    //   [2 MB)   hb0 bf16 : 0.5 MB
    //   [2.5MB)  hb1 bf16 : 0.5 MB
    //   [3 MB)   Wt bf16  : 3 MB   (3072 x 512)
    //   [6 MB)   Ut bf16  : 6 MB   (3072 x 1024)
    char* ws = (char*)d_ws;
    float* h0  = (float*)(ws);
    float* h1  = (float*)(ws + (1u << 20));
    u16*   hb0 = (u16*)  (ws + (2u << 20));
    u16*   hb1 = (u16*)  (ws + (2u << 20) + (1u << 19));
    u16*   Wt  = (u16*)  (ws + (3u << 20));
    u16*   Ut  = (u16*)  (ws + (6u << 20));

    // prep: bf16 transposed weights + zeroed initial state
    gru_cvt_tr_75539884802374<<<(G3 * I_SZ) / 256, 256, 0, stream>>>(wk, Wt, I_SZ);
    gru_cvt_tr_75539884802374<<<(G3 * U_SZ) / 256, 256, 0, stream>>>(rk, Ut, U_SZ);
    gru_zero_u32_75539884802374<<<(B_SZ * U_SZ) / 256, 256, 0, stream>>>(
        (unsigned int*)h0, B_SZ * U_SZ);
    gru_zero_u32_75539884802374<<<(B_SZ * U_SZ / 2) / 256, 256, 0, stream>>>(
        (unsigned int*)hb0, B_SZ * U_SZ / 2);

    dim3 grid(U_SZ / WG_UNITS, B_SZ / WG_ROWS);  // (64, 2) = 128 workgroups
    for (int t = 0; t < T_SZ; ++t) {
        const bool odd = (t & 1) != 0;
        const float* hp  = odd ? h1  : h0;
        const u16*   hpb = odd ? hb1 : hb0;
        float*       hn  = (t == T_SZ - 1) ? out : (odd ? h0 : h1);
        u16*         hnb = odd ? hb0 : hb1;
        gru_step_75539884802374<<<grid, 256, 0, stream>>>(
            x + (long long)t * I_SZ, Wt, Ut, bias, hp, hpb, hn, hnb);
    }
}
